// ActorGNN_72808285602064
// MI455X (gfx1250) — matrix operand order repaired and checked by reading, not compiled, not run on hardware
//
#include <hip/hip_runtime.h>

// ---------------- problem constants ----------------
#define NN      100000      // nodes
#define NE      1600000     // edges
#define IN_DIM  64
#define EDGE_DIM 16
#define HDIM    32
#define ODIM    64
#define NGRAPH  64
#define ACT     18
#define GEN_EPS 1e-7f
#define BN_EPS  1e-5f

// workspace layout (float offsets)
#define SZ_NH   (NN * HDIM)          // 3,200,000
#define O_H1    0
#define O_MX    (1 * SZ_NH)
#define O_DEN   (2 * SZ_NH)
#define O_NUM   (3 * SZ_NH)
#define O_HRES  (4 * SZ_NH)
#define O_YA    0                    // overlaps h1+mx (dead by then)
#define O_YB    (2 * SZ_NH)          // overlaps den+num (dead by then)
#define O_STATS (5 * SZ_NH)          // 16,000,000
// stats sublayout
#define S_BSUM   0                   // 3 layers * 64
#define S_BSQ    192
#define S_POOL   384                 // 64*64
#define S_CNT    4480                // 64
#define S_TOTAL  4544

typedef __attribute__((ext_vector_type(16))) _Float16 v16h;
typedef __attribute__((ext_vector_type(8)))  float    v8f;

union Frag { v16h v; _Float16 e[16]; };

// ---------------- utility ----------------
__global__ void k_zero(float* __restrict__ p, int n) {
    int i = blockIdx.x * 256 + threadIdx.x;
    if (i < n) p[i] = 0.0f;
}

// ---------------- WMMA GEMM:  Y[NN,NOUT] = act(A)[NN,KIN] @ W[NOUT,KIN]^T + bias ----
// Optional fused input transform: BN(scale/shift from column sums) + ReLU.
template<int KIN, int NOUT, bool BN>
__global__ __launch_bounds__(256)
void k_gemm(const float* __restrict__ A, const float* __restrict__ W,
            const float* __restrict__ bias,
            const float* __restrict__ bn_sum, const float* __restrict__ bn_sq,
            const float* __restrict__ gamma,  const float* __restrict__ beta,
            float* __restrict__ Y)
{
    constexpr int KSTEPS = KIN / 32;
    constexpr int NT     = NOUT / 16;
    __shared__ float s_scale[KIN];
    __shared__ float s_shift[KIN];

    int t = threadIdx.x;
    if (BN) {
        if (t < KIN) {
            float mean = bn_sum[t] * (1.0f / (float)NN);
            float var  = bn_sq[t]  * (1.0f / (float)NN) - mean * mean;
            float sc   = gamma[t] * rsqrtf(var + BN_EPS);
            s_scale[t] = sc;
            s_shift[t] = beta[t] - mean * sc;
        }
        __syncthreads();
    }

    int wave = t >> 5;
    int lane = t & 31;
    int tile = blockIdx.x * 8 + wave;
    if (tile >= NN / 16) return;          // uniform per wave; EXEC stays all-1

    int m0     = tile * 16;
    int mrow   = m0 + (lane & 15);
    int aoff   = (lane >> 4) * 16;        // A frag: upper half-lanes hold K+16
    int boff   = (lane >> 4) * 16;        // B frag: upper half-lanes hold K+16
    int ncol_l = lane & 15;

    v8f acc[NT] = {};

    for (int ks = 0; ks < KSTEPS; ks++) {
        Frag a;
        const float* arow = A + (size_t)mrow * KIN + ks * 32 + aoff;
        #pragma unroll
        for (int j = 0; j < 8; j++) {
            float v0 = arow[j];
            float v1 = arow[8 + j];
            if (BN) {
                int k0 = ks * 32 + aoff + j;
                int k1 = k0 + 8;
                v0 = fmaxf(v0 * s_scale[k0] + s_shift[k0], 0.0f);
                v1 = fmaxf(v1 * s_scale[k1] + s_shift[k1], 0.0f);
            }
            a.e[j]     = (_Float16)v0;
            a.e[j + 8] = (_Float16)v1;
        }
        #pragma unroll
        for (int n = 0; n < NT; n++) {
            Frag b;
            const float* wrow = W + (size_t)(n * 16 + ncol_l) * KIN + ks * 32 + boff;
            #pragma unroll
            for (int j = 0; j < 16; j++) b.e[j] = (_Float16)wrow[j];
            acc[n] = __builtin_amdgcn_wmma_f32_16x16x32_f16(
                false, a.v, false, b.v, (short)0, acc[n], false, false);
        }
    }

    #pragma unroll
    for (int n = 0; n < NT; n++) {
        int ncol = n * 16 + ncol_l;
        float bv = bias[ncol];
        int mr0 = m0 + (lane >> 4) * 8;
        #pragma unroll
        for (int r = 0; r < 8; r++)
            Y[(size_t)(mr0 + r) * NOUT + ncol] = acc[n][r] + bv;
    }
}

// ---------------- edge passes: one wave per edge, lane == channel --------------
// msg = relu(h[src] + edge_attr @ edgeW^T + edge_b) + eps   (recomputed, never stored)

__global__ __launch_bounds__(256)
void k_edge_max(const int* __restrict__ ei, const float* __restrict__ eattr,
                const float* __restrict__ eW, const float* __restrict__ eB,
                const float* __restrict__ h1, float* __restrict__ mx)
{
    __shared__ float s_w[HDIM * EDGE_DIM];
    __shared__ float s_b[HDIM];
    int t = threadIdx.x;
    for (int i = t; i < HDIM * EDGE_DIM; i += 256) s_w[i] = eW[i];
    if (t < HDIM) s_b[t] = eB[t];
    __syncthreads();

    int e = blockIdx.x * 8 + (t >> 5);
    if (e >= NE) return;
    int lane = t & 31;
    int src = ei[e];
    int dst = ei[NE + e];
    float acc = s_b[lane];
    const float* ap = eattr + e * EDGE_DIM;
    #pragma unroll
    for (int j = 0; j < EDGE_DIM; j++) acc += ap[j] * s_w[lane * EDGE_DIM + j];
    float msg = fmaxf(h1[src * HDIM + lane] + acc, 0.0f) + GEN_EPS;
    // msg > 0 always -> bit-pattern-monotone uint atomic max (init 0.0f)
    atomicMax((unsigned int*)(mx + dst * HDIM + lane), __float_as_uint(msg));
}

__global__ __launch_bounds__(256)
void k_edge_sum(const int* __restrict__ ei, const float* __restrict__ eattr,
                const float* __restrict__ eW, const float* __restrict__ eB,
                const float* __restrict__ h1, const float* __restrict__ mx,
                float* __restrict__ den, float* __restrict__ num)
{
    __shared__ float s_w[HDIM * EDGE_DIM];
    __shared__ float s_b[HDIM];
    int t = threadIdx.x;
    for (int i = t; i < HDIM * EDGE_DIM; i += 256) s_w[i] = eW[i];
    if (t < HDIM) s_b[t] = eB[t];
    __syncthreads();

    int e = blockIdx.x * 8 + (t >> 5);
    if (e >= NE) return;
    int lane = t & 31;
    int src = ei[e];
    int dst = ei[NE + e];
    float acc = s_b[lane];
    const float* ap = eattr + e * EDGE_DIM;
    #pragma unroll
    for (int j = 0; j < EDGE_DIM; j++) acc += ap[j] * s_w[lane * EDGE_DIM + j];
    float msg = fmaxf(h1[src * HDIM + lane] + acc, 0.0f) + GEN_EPS;
    float w = __expf(msg - mx[dst * HDIM + lane]);
    atomicAdd(den + dst * HDIM + lane, w);
    atomicAdd(num + dst * HDIM + lane, w * msg);
}

// out = softmax-agg + residual:  hres = num/den + h1   (den==0 -> no incoming edges)
__global__ void k_finalize(const float* __restrict__ num, const float* __restrict__ den,
                           const float* __restrict__ h1, float* __restrict__ hres)
{
    int i = blockIdx.x * 256 + threadIdx.x;
    if (i >= NN * HDIM) return;
    float d = den[i];
    float agg = (d > 0.0f) ? num[i] / d : 0.0f;
    hres[i] = agg + h1[i];
}

// per-column sum / sumsq for train-mode BatchNorm
__global__ __launch_bounds__(256)
void k_bnstats(const float* __restrict__ Y, float* __restrict__ sum, float* __restrict__ sq)
{
    int t = blockIdx.x * 256 + threadIdx.x;
    int c = t & 63;
    int r0 = t >> 6;
    int rstride = (gridDim.x * 256) >> 6;
    float s = 0.0f, q = 0.0f;
    for (int r = r0; r < NN; r += rstride) {
        float v = Y[(size_t)r * 64 + c];
        s += v;
        q += v * v;
    }
    atomicAdd(sum + c, s);
    atomicAdd(sq + c, q);
}

// global mean pool (numerator + counts via atomics; batch is sorted -> good locality)
__global__ void k_pool(const float* __restrict__ Y, const int* __restrict__ batch,
                       float* __restrict__ pooled, float* __restrict__ cnt)
{
    int i = blockIdx.x * 256 + threadIdx.x;
    if (i >= NN * 64) return;
    int n = i >> 6, c = i & 63;
    int g = batch[n];
    atomicAdd(pooled + g * 64 + c, Y[i]);
    if (c == 0) atomicAdd(cnt + g, 1.0f);
}

// tiny actor head: [64,64] -> relu [64,16] -> tanh [64,18]
__global__ void k_head(const float* __restrict__ pooled, const float* __restrict__ cnt,
                       const float* __restrict__ W1, const float* __restrict__ b1,
                       const float* __restrict__ W2, const float* __restrict__ b2,
                       float* __restrict__ out)
{
    int g = threadIdx.x;
    if (g >= NGRAPH) return;
    float c = fmaxf(cnt[g], 1.0f);
    float pr[64];
    #pragma unroll
    for (int j = 0; j < 64; j++) pr[j] = pooled[g * 64 + j] / c;
    float a1[16];
    #pragma unroll
    for (int o = 0; o < 16; o++) {
        float s = b1[o];
        for (int j = 0; j < 64; j++) s += pr[j] * W1[o * 64 + j];
        a1[o] = fmaxf(s, 0.0f);
    }
    #pragma unroll
    for (int o = 0; o < ACT; o++) {
        float s = b2[o];
        for (int j = 0; j < 16; j++) s += a1[j] * W2[o * 16 + j];
        out[g * ACT + o] = tanhf(s);
    }
}

// ---------------- launcher ----------------
extern "C" void kernel_launch(void* const* d_in, const int* in_sizes, int n_in,
                              void* d_out, int out_size, void* d_ws, size_t ws_size,
                              hipStream_t stream)
{
    const float* x     = (const float*)d_in[0];
    const int*   ei    = (const int*)  d_in[1];
    const float* eattr = (const float*)d_in[2];
    const int*   batch = (const int*)  d_in[3];
    const float* nodeW = (const float*)d_in[4];
    const float* nodeB = (const float*)d_in[5];
    const float* edgeW = (const float*)d_in[6];
    const float* edgeB = (const float*)d_in[7];
    const float* W0 = (const float*)d_in[8];
    const float* W1 = (const float*)d_in[9];
    const float* W2 = (const float*)d_in[10];
    const float* W3 = (const float*)d_in[11];
    const float* B0 = (const float*)d_in[12];
    const float* B1 = (const float*)d_in[13];
    const float* B2 = (const float*)d_in[14];
    const float* B3 = (const float*)d_in[15];
    const float* G0 = (const float*)d_in[16];
    const float* G1 = (const float*)d_in[17];
    const float* G2 = (const float*)d_in[18];
    const float* E0 = (const float*)d_in[19];
    const float* E1 = (const float*)d_in[20];
    const float* E2 = (const float*)d_in[21];
    const float* aiW = (const float*)d_in[22];
    const float* aiB = (const float*)d_in[23];
    const float* aoW = (const float*)d_in[24];
    const float* aoB = (const float*)d_in[25];

    float* ws   = (float*)d_ws;
    float* h1   = ws + O_H1;
    float* mx   = ws + O_MX;
    float* den  = ws + O_DEN;
    float* num  = ws + O_NUM;
    float* hres = ws + O_HRES;
    float* Ya   = ws + O_YA;
    float* Yb   = ws + O_YB;
    float* bsum = ws + O_STATS + S_BSUM;
    float* bsq  = ws + O_STATS + S_BSQ;
    float* pooled = ws + O_STATS + S_POOL;
    float* cnt    = ws + O_STATS + S_CNT;

    const int GB = (NN / 16 + 7) / 8;      // 782 blocks of 8 waves for GEMMs

    // zero accumulators (mx, den, num contiguous) + stats region
    k_zero<<<(3 * SZ_NH + 255) / 256, 256, 0, stream>>>(ws + O_MX, 3 * SZ_NH);
    k_zero<<<(S_TOTAL + 255) / 256, 256, 0, stream>>>(ws + O_STATS, S_TOTAL);

    // node encoder: h1 = x @ nodeW^T + nodeB
    k_gemm<64, 32, false><<<GB, 256, 0, stream>>>(x, nodeW, nodeB,
        nullptr, nullptr, nullptr, nullptr, h1);

    // GENConv softmax aggregation (two fused edge passes)
    k_edge_max<<<NE / 8, 256, 0, stream>>>(ei, eattr, edgeW, edgeB, h1, mx);
    k_edge_sum<<<NE / 8, 256, 0, stream>>>(ei, eattr, edgeW, edgeB, h1, mx, den, num);
    k_finalize<<<(NN * HDIM + 255) / 256, 256, 0, stream>>>(num, den, h1, hres);

    // MLP: Lin -> BN -> ReLU (x3) -> Lin, BN fused into next layer's A-load
    k_gemm<32, 64, false><<<GB, 256, 0, stream>>>(hres, W0, B0,
        nullptr, nullptr, nullptr, nullptr, Ya);
    k_bnstats<<<1024, 256, 0, stream>>>(Ya, bsum + 0,   bsq + 0);
    k_gemm<64, 64, true><<<GB, 256, 0, stream>>>(Ya, W1, B1,
        bsum + 0,   bsq + 0,   G0, E0, Yb);
    k_bnstats<<<1024, 256, 0, stream>>>(Yb, bsum + 64,  bsq + 64);
    k_gemm<64, 64, true><<<GB, 256, 0, stream>>>(Yb, W2, B2,
        bsum + 64,  bsq + 64,  G1, E1, Ya);
    k_bnstats<<<1024, 256, 0, stream>>>(Ya, bsum + 128, bsq + 128);
    k_gemm<64, 64, true><<<GB, 256, 0, stream>>>(Ya, W3, B3,
        bsum + 128, bsq + 128, G2, E2, Yb);

    // pool + head
    k_pool<<<(NN * 64 + 255) / 256, 256, 0, stream>>>(Yb, batch, pooled, cnt);
    k_head<<<1, 64, 0, stream>>>(pooled, cnt, aiW, aiB, aoW, aoB, (float*)d_out);
}
